// AdaProj_37142877176475
// MI455X (gfx1250) — matrix operand order repaired and checked by reading, not compiled, not run on hardware
//
#include <hip/hip_runtime.h>
#include <hip/hip_fp16.h>

// ---------------------------------------------------------------------------
// AdaProj, algebraically reduced:
//   out[b,c] = (sum_s l_s^2) / max(sqrt(sum_{s,s'} l_s l_s' G[c,s,s']), eps)
// where l_s = xn[b]·Wn[c,s], G[c] = Wn[c] Wn[c]^T (4x4).
// Only real work: logits GEMM  [256,512] x [4000,512]^T  -> WMMA f16->f32.
// GEMM kernel processes 2 N-tiles per wave so the A fragment is loaded once
// per K-step and reused by two v_wmma issues (3 b128 loads per WMMA).
// ---------------------------------------------------------------------------

typedef __attribute__((ext_vector_type(16))) _Float16 v16h;
typedef __attribute__((ext_vector_type(8)))  _Float16 v8h;
typedef __attribute__((ext_vector_type(8)))  float    v8f;

#define BB   256
#define CC   1000
#define SS   4
#define DD   512
#define EPSF 1e-12f

__device__ __forceinline__ float wave_reduce_add(float v) {
    // wave32 butterfly reduction
    #pragma unroll
    for (int off = 16; off > 0; off >>= 1)
        v += __shfl_xor(v, off, 32);
    return v;
}

// --- Kernel 1: L2-normalize x rows, emit f16 --------------------------------
__global__ void adaproj_norm_x(const float* __restrict__ x,
                               _Float16* __restrict__ xn) {
    const int row  = blockIdx.x;     // 0..255
    const int lane = threadIdx.x;    // 0..31
    const float4* xr = (const float4*)(x + row * DD) + lane * 4;  // 16 floats/lane
    float v[16];
    float ss = 0.f;
    #pragma unroll
    for (int q = 0; q < 4; ++q) {
        float4 f = xr[q];
        v[q*4+0] = f.x; v[q*4+1] = f.y; v[q*4+2] = f.z; v[q*4+3] = f.w;
        ss += f.x*f.x + f.y*f.y + f.z*f.z + f.w*f.w;
    }
    ss = wave_reduce_add(ss);
    const float scale = 1.0f / fmaxf(sqrtf(ss), EPSF);
    _Float16* orow = xn + row * DD + lane * 16;
    #pragma unroll
    for (int i = 0; i < 16; ++i)
        orow[i] = (_Float16)(v[i] * scale);
}

// --- Kernel 2: L2-normalize W rows (per class), emit f16 Wn + f32 Gram ------
__global__ void adaproj_norm_w(const float* __restrict__ W,
                               _Float16* __restrict__ Wn,
                               float* __restrict__ G) {
    const int c    = blockIdx.x;     // 0..999
    const int lane = threadIdx.x;    // 0..31
    float r[SS][16];                 // normalized rows kept in registers
    #pragma unroll
    for (int s = 0; s < SS; ++s) {
        const float4* wr = (const float4*)(W + (c * SS + s) * DD) + lane * 4;
        float ss = 0.f;
        #pragma unroll
        for (int q = 0; q < 4; ++q) {
            float4 f = wr[q];
            r[s][q*4+0] = f.x; r[s][q*4+1] = f.y;
            r[s][q*4+2] = f.z; r[s][q*4+3] = f.w;
            ss += f.x*f.x + f.y*f.y + f.z*f.z + f.w*f.w;
        }
        ss = wave_reduce_add(ss);
        const float scale = 1.0f / fmaxf(sqrtf(ss), EPSF);
        _Float16* orow = Wn + (c * SS + s) * DD + lane * 16;
        #pragma unroll
        for (int i = 0; i < 16; ++i) {
            r[s][i] *= scale;
            orow[i] = (_Float16)r[s][i];
        }
    }
    // Gram matrix G[c] = Wn[c] Wn[c]^T (4x4, symmetric)
    #pragma unroll
    for (int s = 0; s < SS; ++s) {
        #pragma unroll
        for (int t = s; t < SS; ++t) {
            float p = 0.f;
            #pragma unroll
            for (int i = 0; i < 16; ++i) p += r[s][i] * r[t][i];
            p = wave_reduce_add(p);
            if (lane == 0) {
                G[c * 16 + s * 4 + t] = p;
                G[c * 16 + t * 4 + s] = p;
            }
        }
    }
}

// --- Kernel 3: WMMA logits GEMM (2 N-tiles/wave) + fused Gram epilogue ------
// grid = (B/16, C/8); block = 32 (one wave). Each wave: 16x32 (b x (c,s))
// logits tile via 32 x v_wmma_f32_16x16x32_f16 (A fragment reused by both
// N-tiles each K-step), then 16x8 outputs.
__global__ void adaproj_gemm_out(const _Float16* __restrict__ xn,
                                 const _Float16* __restrict__ Wn,
                                 const float* __restrict__ G,
                                 float* __restrict__ out) {
    __shared__ float tile[16][33];        // 16 rows x 32 (c,s) cols, padded

    const int lane  = threadIdx.x;        // 0..31
    const int btile = blockIdx.x * 16;    // batch-row base
    const int ctile = blockIdx.y * 8;     // class base (8 classes = 32 cols)
    const int col0  = blockIdx.y * 32;    // (c,s)-flattened column base

    const int mr = lane & 15;
    const int hi = lane >> 4;

    // A fragment (contiguous-K per lane): lane<16 -> row mr, K 0..15;
    // lane>=16 -> row mr, K 16..31  => two b128 loads, offset hi*16.
    const _Float16* arow  = xn + (btile + mr) * DD + hi * 16;
    // B fragments (column = Wn row, contiguous-K per lane): lane<16 -> col mr,
    // K 0..15; lane>=16 -> col mr, K 16..31  => two b128 loads, offset hi*16.
    const _Float16* brow0 = Wn + (col0 + mr) * DD + hi * 16;
    const _Float16* brow1 = brow0 + 16 * DD;   // second 16-column tile

    union Frag { v16h v; v8h h[2]; };
    v8f acc0 = {};
    v8f acc1 = {};
    #pragma unroll
    for (int kk = 0; kk < DD / 32; ++kk) {
        const int k0 = kk * 32;
        Frag a, b0, b1;
        a.h[0]  = *(const v8h*)(arow  + k0);
        a.h[1]  = *(const v8h*)(arow  + k0 + 8);
        b0.h[0] = *(const v8h*)(brow0 + k0);
        b0.h[1] = *(const v8h*)(brow0 + k0 + 8);
        b1.h[0] = *(const v8h*)(brow1 + k0);
        b1.h[1] = *(const v8h*)(brow1 + k0 + 8);
        acc0 = __builtin_amdgcn_wmma_f32_16x16x32_f16(
            false, a.v, false, b0.v, (short)0, acc0, false, false);
        acc1 = __builtin_amdgcn_wmma_f32_16x16x32_f16(
            false, a.v, false, b1.v, (short)0, acc1, false, false);
    }

    // C/D layout: lane holds column N=mr, rows M = r + 8*hi (r=0..7).
    #pragma unroll
    for (int r = 0; r < 8; ++r) {
        tile[r + 8 * hi][mr]      = acc0[r];
        tile[r + 8 * hi][16 + mr] = acc1[r];
    }
    __syncthreads();

    // 128 outputs per block (16 b-rows x 8 classes): 4 per lane.
    #pragma unroll
    for (int j = 0; j < 4; ++j) {
        const int o  = lane + j * 32;
        const int m  = o & 15;
        const int cl = o >> 4;            // 0..7
        float l[SS];
        #pragma unroll
        for (int s = 0; s < SS; ++s) l[s] = tile[m][cl * 4 + s];
        const float* g = G + (ctile + cl) * 16;
        float num = 0.f, den = 0.f;
        #pragma unroll
        for (int s = 0; s < SS; ++s) {
            num += l[s] * l[s];
            #pragma unroll
            for (int t = 0; t < SS; ++t)
                den += l[s] * l[t] * g[s * 4 + t];
        }
        out[(btile + m) * CC + (ctile + cl)] = num / fmaxf(sqrtf(den), EPSF);
    }
}

// ---------------------------------------------------------------------------
extern "C" void kernel_launch(void* const* d_in, const int* in_sizes, int n_in,
                              void* d_out, int out_size, void* d_ws, size_t ws_size,
                              hipStream_t stream) {
    (void)in_sizes; (void)n_in; (void)out_size; (void)ws_size;
    const float* x = (const float*)d_in[0];      // [256, 512]
    const float* W = (const float*)d_in[1];      // [1000, 4, 512]
    float* out = (float*)d_out;                  // [256, 1000]

    // Workspace layout (total ~4.42 MB)
    char* ws = (char*)d_ws;
    _Float16* xn = (_Float16*)ws;                              // 256*512*2   = 256 KB
    _Float16* Wn = (_Float16*)(ws + (size_t)BB * DD * 2);      // 4000*512*2  = 4 MB
    float*    G  = (float*)(ws + (size_t)BB * DD * 2
                               + (size_t)CC * SS * DD * 2);    // 1000*16*4   = 64 KB

    adaproj_norm_x<<<BB, 32, 0, stream>>>(x, xn);
    adaproj_norm_w<<<CC, 32, 0, stream>>>(W, Wn, G);
    adaproj_gemm_out<<<dim3(BB / 16, CC / 8), 32, 0, stream>>>(xn, Wn, G, out);
}
